// GPTAve_53489522704418
// MI455X (gfx1250) — matrix operand order repaired and checked by reading, not compiled, not run on hardware
//
#include <hip/hip_runtime.h>
#include <hip/hip_bf16.h>
#include <math.h>

#define L_LAYERS 4
#define DMODEL   1024
#define NHEAD    16
#define DHEAD    64
#define SEQ      2048
#define VOCAB    32000
#define DFF      4096
#define NEGINF   (-1.0e9f)
#define LNEPS    1e-5f

typedef __bf16 bf16;
typedef __attribute__((ext_vector_type(16))) __bf16 v16bf;
typedef __attribute__((ext_vector_type(8)))  __bf16 v8bf;
typedef __attribute__((ext_vector_type(4)))  __bf16 v4bf;
typedef __attribute__((ext_vector_type(8)))  float  v8f;
typedef __attribute__((ext_vector_type(4)))  float  v4f;
typedef __attribute__((ext_vector_type(4)))  unsigned int v4u;
typedef __attribute__((ext_vector_type(8)))  int v8i;
typedef __attribute__((ext_vector_type(4)))  int v4i;

#if __has_builtin(__builtin_amdgcn_tensor_load_to_lds) && __has_builtin(__builtin_amdgcn_s_wait_tensorcnt)
#define USE_TDM 1
#else
#define USE_TDM 0
#endif

// ---------------- GEMM configuration ----------------
constexpr int BM = 128;        // block tile M
constexpr int BN = 256;        // block tile N
constexpr int KT = 32;         // K step (one wmma K)
constexpr int LDK = KT + 8;    // LDS row stride in bf16 elems (80B, 16B aligned)
constexpr int NTHREADS = 256;  // 8 wave32

__device__ __forceinline__ v8f zero_v8f() {
  v8f z;
#pragma unroll
  for (int i = 0; i < 8; ++i) z[i] = 0.0f;
  return z;
}

#if USE_TDM
// TDM: DMA a 2D bf16 tile (rows x 32 elems, row stride = stride elems) into LDS
// with 16B padding after each 64B row -> LDS row stride 80B (= LDK bf16).
// Descriptor packing per CDNA5 ISA sec. 8.3 - 8.6. 6-arg builtin form
// (g0, g1, g2, g3, <extra group, zeroed>, cpol).
__device__ __forceinline__ void tdm_load_2d(unsigned int lds_addr, const void* gptr,
                                            int rows, int stride_elems) {
  unsigned long long ga = (unsigned long long)gptr;
  v4u g0;
  g0[0] = 1u;                                   // count=1, user mode
  g0[1] = lds_addr;                             // LDS byte address
  g0[2] = (unsigned int)ga;                     // global addr [31:0]
  g0[3] = (unsigned int)((ga >> 32) & 0x01FFFFFFu) | (2u << 30);  // addr[56:32] | type=2
  v8i g1;
  g1[0] = (int)((1u << 16)      // data_size = 2 bytes
              | (1u << 20)      // pad_enable
              | (3u << 22)      // pad_interval: 16 DWORDs (64B)
              | (3u << 25));    // pad_amount: 4 DWORDs (16B)
  g1[1] = 0;                                        // tensor_dim0[15:0]=0 (dim0 = 1<<30)
  g1[2] = (int)0x4000;                              // tensor_dim0[31:16] ; tensor_dim1[15:0]=0
  g1[3] = (int)(0x4000u | ((unsigned)KT << 16));    // tensor_dim1[31:16] ; tile_dim0=32
  g1[4] = rows;                                     // tile_dim1 ; tile_dim2=0 (2D)
  g1[5] = stride_elems;                             // tensor_dim0_stride[31:0]
  g1[6] = 0;                                        // stride[47:32] ; tensor_dim1_stride lo
  g1[7] = 0;
  v4i g2;
  g2[0] = (int)0x40000000;  // tensor_dim2 (unused, large)
  g2[1] = (int)0x40000000;  // tensor_dim3 (unused, large)
  g2[2] = 0;                // tensor_dim2_stride lo
  g2[3] = 0;                // stride hi | tile_dim3=0
  v4i g3;
  g3[0] = 0;                // tensor_dim3_stride lo
  g3[1] = 0;                // stride hi | tensor_dim4 lo
  g3[2] = (int)0x4000;      // tensor_dim4 hi | tile_dim4=0
  g3[3] = 0;
  v8i gx;
#pragma unroll
  for (int i = 0; i < 8; ++i) gx[i] = 0;
  __builtin_amdgcn_tensor_load_to_lds(g0, g1, g2, g3, gx, 0);
}
#endif

// A fragment (16x32 bf16): lanes 0-15 row M=lane, K 0..7 / 16..23;
// lanes 16-31 row M=lane-16, K 8..15 / 24..31.  (ISA 7.12.2)
__device__ __forceinline__ v16bf load_fragA(const bf16* tile, int rowBase, int lane) {
  const int m = lane & 15, half = lane >> 4;
  const bf16* r = tile + (rowBase + m) * LDK;
  v8bf lo = *(const v8bf*)(r + half * 8);
  v8bf hs = *(const v8bf*)(r + 16 + half * 8);
  v16bf out;
#pragma unroll
  for (int e = 0; e < 8; ++e) { out[e] = lo[e]; out[e + 8] = hs[e]; }
  return out;
}

// B fragment (32x16 bf16, stored transposed [n][k] in LDS): lane holds column
// n=lane&15; lanes 0-15 K 0..7 / 16..23, lanes 16-31 K 8..15 / 24..31.
__device__ __forceinline__ v16bf load_fragB(const bf16* tile, int colBase, int lane) {
  const int n = lane & 15, half = lane >> 4;
  const bf16* r = tile + (colBase + n) * LDK;
  v8bf lo = *(const v8bf*)(r + half * 8);
  v8bf hs = *(const v8bf*)(r + 16 + half * 8);
  v16bf out;
#pragma unroll
  for (int e = 0; e < 8; ++e) { out[e] = lo[e]; out[e + 8] = hs[e]; }
  return out;
}

// EPI: 0 = none, 1 = +bias, 2 = +bias then exact GELU, 3 = attention mask
// TB:  float (convert to bf16 on LDS stage) or unsigned short (bf16 bits)
// BT:  false -> B is [K,N] row-major; true -> B is [N,K] row-major (A*B^T)
template <typename TB, bool BT, int EPI, bool OUTBF>
__global__ __launch_bounds__(NTHREADS)
void gemm_wmma(const unsigned short* A_, long aZ, int lda,
               const void* B_, long bZ, int ldb,
               void* C_, long cZ, int ldc,
               const float* bias, int M, int N, int K) {
  __shared__ bf16 As[2][BM * LDK];
  __shared__ bf16 Bs[2][BN * LDK];

  const int tid  = threadIdx.x;
  const int lane = tid & 31;
  const int wid  = tid >> 5;
  const int wm   = (wid >> 2) * 64;  // wave M offset: 0 / 64
  const int wn   = (wid & 3) * 64;   // wave N offset: 0/64/128/192
  const int m0   = blockIdx.y * BM;
  const int n0   = blockIdx.x * BN;
  const int z    = blockIdx.z;

  const bf16* A  = (const bf16*)A_ + (long)z * aZ;
  const TB*   Bp = (const TB*)B_ + (long)z * bZ;

  // TDM handles: A tile always (bf16); B tile when bf16 and B-transposed (QK^T).
  constexpr bool TDMB = (USE_TDM != 0) && BT && (sizeof(TB) == 2);

  v4f  bregf[8];   // fp32 B staging regs
  v8bf bregh[4];   // bf16 B staging regs
#if !USE_TDM
  v8bf aregs[2];   // A staging regs (fallback)
#endif

  auto stageA_issue = [&](int k0, int buf) {
#if USE_TDM
    if (wid == 0)
      tdm_load_2d((unsigned int)(unsigned long long)&As[buf][0],
                  (const unsigned char*)A + ((long)m0 * lda + (long)k0) * 2, BM, lda);
#else
    (void)buf;
#pragma unroll
    for (int t = 0; t < 2; ++t) {
      int v = tid + t * NTHREADS;
      int row = v >> 2, kc = (v & 3) * 8;
      aregs[t] = *(const v8bf*)(A + (long)(m0 + row) * lda + k0 + kc);
    }
#endif
  };
  auto stageA_store = [&](int buf) {
#if !USE_TDM
#pragma unroll
    for (int t = 0; t < 2; ++t) {
      int v = tid + t * NTHREADS;
      int row = v >> 2, kc = (v & 3) * 8;
      *(v8bf*)&As[buf][row * LDK + kc] = aregs[t];
    }
#else
    (void)buf;
#endif
  };

  auto stageB_issue = [&](int k0, int buf) {
    if constexpr (TDMB) {
#if USE_TDM
      if (wid == 0)
        tdm_load_2d((unsigned int)(unsigned long long)&Bs[buf][0],
                    (const unsigned char*)Bp + ((long)n0 * ldb + (long)k0) * 2, BN, ldb);
#endif
      (void)buf;
    } else if constexpr (!BT && sizeof(TB) == 4) {   // fp32 [K,N]
#pragma unroll
      for (int t = 0; t < 8; ++t) {
        int v = tid + t * NTHREADS;
        int k = v >> 6, n = (v & 63) * 4;
        v4f f;
#pragma unroll
        for (int e = 0; e < 4; ++e) f[e] = 0.0f;
        if (n0 + n < N)
          f = *(const v4f*)((const float*)Bp + (long)(k0 + k) * ldb + n0 + n);
        bregf[t] = f;
      }
    } else if constexpr (!BT && sizeof(TB) == 2) {   // bf16 [K,N]
#pragma unroll
      for (int t = 0; t < 4; ++t) {
        int v = tid + t * NTHREADS;
        int k = v >> 5, n = (v & 31) * 8;
        v8bf f;
#pragma unroll
        for (int e = 0; e < 8; ++e) f[e] = (bf16)0.0f;
        if (n0 + n < N)
          f = *(const v8bf*)((const bf16*)Bp + (long)(k0 + k) * ldb + n0 + n);
        bregh[t] = f;
      }
    } else if constexpr (BT && sizeof(TB) == 4) {    // fp32 [N,K]
#pragma unroll
      for (int t = 0; t < 8; ++t) {
        int v = tid + t * NTHREADS;
        int n = v >> 3, k = (v & 7) * 4;
        v4f f;
#pragma unroll
        for (int e = 0; e < 4; ++e) f[e] = 0.0f;
        if (n0 + n < N)
          f = *(const v4f*)((const float*)Bp + (long)(n0 + n) * ldb + k0 + k);
        bregf[t] = f;
      }
    } else {                                          // bf16 [N,K] (no TDM build)
#pragma unroll
      for (int t = 0; t < 4; ++t) {
        int v = tid + t * NTHREADS;
        int n = v >> 2, k = (v & 3) * 8;
        v8bf f;
#pragma unroll
        for (int e = 0; e < 8; ++e) f[e] = (bf16)0.0f;
        if (n0 + n < N)
          f = *(const v8bf*)((const bf16*)Bp + (long)(n0 + n) * ldb + k0 + k);
        bregh[t] = f;
      }
    }
  };

  auto stageB_store = [&](int buf) {
    if constexpr (TDMB) {
      (void)buf;
    } else if constexpr (!BT && sizeof(TB) == 4) {
#pragma unroll
      for (int t = 0; t < 8; ++t) {
        int v = tid + t * NTHREADS;
        int k = v >> 6, n = (v & 63) * 4;
#pragma unroll
        for (int e = 0; e < 4; ++e) Bs[buf][(n + e) * LDK + k] = (bf16)bregf[t][e];
      }
    } else if constexpr (!BT && sizeof(TB) == 2) {
#pragma unroll
      for (int t = 0; t < 4; ++t) {
        int v = tid + t * NTHREADS;
        int k = v >> 5, n = (v & 31) * 8;
#pragma unroll
        for (int e = 0; e < 8; ++e) Bs[buf][(n + e) * LDK + k] = bregh[t][e];
      }
    } else if constexpr (BT && sizeof(TB) == 4) {
#pragma unroll
      for (int t = 0; t < 8; ++t) {
        int v = tid + t * NTHREADS;
        int n = v >> 3, k = (v & 7) * 4;
        v4bf cv;
#pragma unroll
        for (int e = 0; e < 4; ++e) cv[e] = (bf16)bregf[t][e];
        *(v4bf*)&Bs[buf][n * LDK + k] = cv;
      }
    } else {
#pragma unroll
      for (int t = 0; t < 4; ++t) {
        int v = tid + t * NTHREADS;
        int n = v >> 2, k = (v & 3) * 8;
        *(v8bf*)&Bs[buf][n * LDK + k] = bregh[t];
      }
    }
  };

  v8f acc[4][4];
#pragma unroll
  for (int i = 0; i < 4; ++i)
#pragma unroll
    for (int j = 0; j < 4; ++j) acc[i][j] = zero_v8f();

  const int nsteps = K / KT;

  // ---- prologue: stage tile 0 into buffer 0
  stageA_issue(0, 0);
  stageB_issue(0, 0);
  stageA_store(0);
  stageB_store(0);
#if USE_TDM
  if (wid == 0) __builtin_amdgcn_s_wait_tensorcnt(0);
#endif
  __syncthreads();

  // ---- double-buffered main loop: prefetch tile s+1 while computing tile s
  for (int s = 0; s < nsteps; ++s) {
    const int cur = s & 1;
    const int nxt = cur ^ 1;
    const bool pre = (s + 1) < nsteps;
    if (pre) {
      stageA_issue((s + 1) * KT, nxt);   // TDM DMA / global loads in flight
      stageB_issue((s + 1) * KT, nxt);
    }
    const bf16* Ac = &As[cur][0];
    const bf16* Bc = &Bs[cur][0];
    v16bf af[4], bfv[4];
#pragma unroll
    for (int i = 0; i < 4; ++i) af[i]  = load_fragA(Ac, wm + i * 16, lane);
#pragma unroll
    for (int j = 0; j < 4; ++j) bfv[j] = load_fragB(Bc, wn + j * 16, lane);
#pragma unroll
    for (int i = 0; i < 4; ++i)
#pragma unroll
      for (int j = 0; j < 4; ++j)
        acc[i][j] = __builtin_amdgcn_wmma_f32_16x16x32_bf16(
            false, af[i], false, bfv[j], (short)0, acc[i][j], false, false);
    if (pre) {
      stageA_store(nxt);
      stageB_store(nxt);
#if USE_TDM
      if (wid == 0) __builtin_amdgcn_s_wait_tensorcnt(0);
#endif
    }
    __syncthreads();
  }

  // ---- epilogue: C layout (ISA 7.12.2): VGPR r -> M = r + 8*(lane>=16), N = lane&15
  const int m16 = lane & 15, half = lane >> 4;
  float* Cf = (float*)C_ + (long)z * cZ;
  bf16*  Cb = (bf16*)C_ + (long)z * cZ;
#pragma unroll
  for (int j = 0; j < 4; ++j) {
    int col = n0 + wn + j * 16 + m16;
    if (col >= N) continue;
    float bv = 0.0f;
    if constexpr (EPI == 1 || EPI == 2) bv = bias[col];
#pragma unroll
    for (int i = 0; i < 4; ++i) {
#pragma unroll
      for (int r = 0; r < 8; ++r) {
        int row = m0 + wm + i * 16 + half * 8 + r;
        float v = acc[i][j][r] + bv;
        if constexpr (EPI == 2)
          v = 0.5f * v * (1.0f + erff(v * 0.70710678118654752f));
        if constexpr (EPI == 3)
          if (row < NHEAD && col > row) v += NEGINF;
        if constexpr (OUTBF) Cb[(long)row * ldc + col] = (bf16)v;
        else                 Cf[(long)row * ldc + col] = v;
      }
    }
  }
}

// ---------------- reductions ----------------
__device__ __forceinline__ float waveSum(float v) {
#pragma unroll
  for (int o = 16; o > 0; o >>= 1) v += __shfl_xor(v, o, 32);
  return v;
}
__device__ __forceinline__ float waveMax(float v) {
#pragma unroll
  for (int o = 16; o > 0; o >>= 1) v = fmaxf(v, __shfl_xor(v, o, 32));
  return v;
}
__device__ __forceinline__ float blockSum(float v, float* red) {
  const int lane = threadIdx.x & 31, wid = threadIdx.x >> 5;
  v = waveSum(v);
  __syncthreads();
  if (lane == 0) red[wid] = v;
  __syncthreads();
  float t = 0.0f;
#pragma unroll
  for (int i = 0; i < 8; ++i) t += red[i];
  return t;
}
__device__ __forceinline__ float blockMax(float v, float* red) {
  const int lane = threadIdx.x & 31, wid = threadIdx.x >> 5;
  v = waveMax(v);
  __syncthreads();
  if (lane == 0) red[wid] = v;
  __syncthreads();
  float t = red[0];
#pragma unroll
  for (int i = 1; i < 8; ++i) t = fmaxf(t, red[i]);
  return t;
}

// ---------------- elementwise kernels ----------------
__global__ __launch_bounds__(256)
void cast_copy_kernel(const float* __restrict__ x, unsigned short* __restrict__ xb,
                      float* __restrict__ xf, int n) {
  int i = blockIdx.x * 256 + threadIdx.x;
  if (i < n) {
    float v = x[i];
    ((bf16*)xb)[i] = (bf16)v;
    xf[i] = v;
  }
}

// x_out = layernorm(a + b) * g + be ; writes bf16 + f32 copies
__global__ __launch_bounds__(256)
void ln_residual_kernel(const float* __restrict__ a, const float* __restrict__ b,
                        const float* __restrict__ g, const float* __restrict__ be,
                        unsigned short* __restrict__ out_bf, float* __restrict__ out_f) {
  __shared__ float red[8];
  const int row = blockIdx.x, tid = threadIdx.x;
  const float* ar = a + (long)row * DMODEL;
  const float* br = b + (long)row * DMODEL;
  float v[4];
  float s = 0.0f;
#pragma unroll
  for (int t = 0; t < 4; ++t) {
    int c = tid + t * 256;
    float x = ar[c] + br[c];
    v[t] = x; s += x;
  }
  s = blockSum(s, red);
  const float mean = s * (1.0f / DMODEL);
  float vs = 0.0f;
#pragma unroll
  for (int t = 0; t < 4; ++t) { float d = v[t] - mean; vs += d * d; }
  vs = blockSum(vs, red);
  const float rstd = rsqrtf(vs * (1.0f / DMODEL) + LNEPS);
  bf16* ob = (bf16*)out_bf;
#pragma unroll
  for (int t = 0; t < 4; ++t) {
    int c = tid + t * 256;
    float o = (v[t] - mean) * rstd * g[c] + be[c];
    out_f[(long)row * DMODEL + c] = o;
    ob[(long)row * DMODEL + c] = (bf16)o;
  }
}

// row softmax over S, f32 scores -> in-place bf16 attn (same byte base per row)
__global__ __launch_bounds__(256)
void softmax_attn_kernel(float* __restrict__ scores) {
  __shared__ float red[8];
  const int srow = blockIdx.x, h = blockIdx.y, tid = threadIdx.x;
  float* rowp = scores + ((long)h * SEQ + srow) * SEQ;
  float v[8];
  float mx = -3.0e38f;
#pragma unroll
  for (int t = 0; t < 8; ++t) {
    float x = rowp[tid + t * 256];
    v[t] = x; mx = fmaxf(mx, x);
  }
  mx = blockMax(mx, red);          // barrier => all reads complete before writes
  float s = 0.0f;
#pragma unroll
  for (int t = 0; t < 8; ++t) { float e = expf(v[t] - mx); v[t] = e; s += e; }
  s = blockSum(s, red);
  const float inv = 1.0f / s;
  bf16* out = (bf16*)rowp;         // bf16 row overlays first half of f32 row
#pragma unroll
  for (int t = 0; t < 8; ++t) out[tid + t * 256] = (bf16)(v[t] * inv);
}

// in-place row softmax over VOCAB (f32)
__global__ __launch_bounds__(256)
void softmax_vocab_kernel(float* __restrict__ logits) {
  __shared__ float red[8];
  const int row = blockIdx.x, tid = threadIdx.x;
  float* rp = logits + (long)row * VOCAB;
  float mx = -3.0e38f;
  for (int c = tid; c < VOCAB; c += 256) mx = fmaxf(mx, rp[c]);
  mx = blockMax(mx, red);
  float s = 0.0f;
  for (int c = tid; c < VOCAB; c += 256) s += expf(rp[c] - mx);
  s = blockSum(s, red);
  const float inv = 1.0f / s;
  for (int c = tid; c < VOCAB; c += 256) rp[c] = expf(rp[c] - mx) * inv;
}

// ---------------- host orchestration ----------------
extern "C" void kernel_launch(void* const* d_in, const int* in_sizes, int n_in,
                              void* d_out, int out_size, void* d_ws, size_t ws_size,
                              hipStream_t stream) {
  (void)in_sizes; (void)n_in; (void)out_size; (void)ws_size;
  const float* x_in = (const float*)d_in[0];
  const float* emb  = (const float*)d_in[1];
  const float* Wq = (const float*)d_in[2];
  const float* Wk = (const float*)d_in[3];
  const float* Wv = (const float*)d_in[4];
  const float* Wp = (const float*)d_in[5];
  const float* bq = (const float*)d_in[6];
  const float* bk = (const float*)d_in[7];
  const float* bv = (const float*)d_in[8];
  const float* bp = (const float*)d_in[9];
  const float* W1 = (const float*)d_in[10];
  const float* b1 = (const float*)d_in[11];
  const float* W2 = (const float*)d_in[12];
  const float* b2 = (const float*)d_in[13];
  const float* g1 = (const float*)d_in[14];
  const float* be1 = (const float*)d_in[15];
  const float* g2 = (const float*)d_in[16];
  const float* be2 = (const float*)d_in[17];

  char* p = (char*)d_ws;
  auto take = [&](size_t bytes) -> void* {
    void* r = (void*)p;
    p += (bytes + 255) & ~(size_t)255;
    return r;
  };
  unsigned short* x_bf  = (unsigned short*)take((size_t)SEQ * DMODEL * 2);
  float*          x_f   = (float*)take((size_t)SEQ * DMODEL * 4);
  unsigned short* q_bf  = (unsigned short*)take((size_t)SEQ * DMODEL * 2);
  unsigned short* k_bf  = (unsigned short*)take((size_t)SEQ * DMODEL * 2);
  unsigned short* v_bf  = (unsigned short*)take((size_t)SEQ * DMODEL * 2);
  unsigned short* oh_bf = (unsigned short*)take((size_t)SEQ * DMODEL * 2);
  float*          oproj = (float*)take((size_t)SEQ * DMODEL * 4);
  unsigned short* x1_bf = (unsigned short*)take((size_t)SEQ * DMODEL * 2);
  float*          x1_f  = (float*)take((size_t)SEQ * DMODEL * 4);
  unsigned short* h1_bf = (unsigned short*)take((size_t)SEQ * DFF * 2);
  float*          y_f   = (float*)take((size_t)SEQ * DMODEL * 4);
  float*          scores = (float*)take((size_t)NHEAD * SEQ * SEQ * 4);

  const dim3 blk(NTHREADS);
  cast_copy_kernel<<<(SEQ * DMODEL + 255) / 256, blk, 0, stream>>>(x_in, x_bf, x_f, SEQ * DMODEL);

  const dim3 gProj(DMODEL / BN, SEQ / BM, 1);       // 4 x 16
  const dim3 gScore(SEQ / BN, SEQ / BM, NHEAD);     // 8 x 16 x 16
  const dim3 gAV(1, SEQ / BM, NHEAD);               // N=64 -> 1 x 16 x 16
  const dim3 gF1(DFF / BN, SEQ / BM, 1);            // 16 x 16
  const dim3 gLog(VOCAB / BN, SEQ / BM, 1);         // 125 x 16
  const dim3 gSm(SEQ, NHEAD);

  for (int i = 0; i < L_LAYERS; ++i) {
    const float* Wq_i = Wq + (size_t)i * DMODEL * DMODEL;
    const float* Wk_i = Wk + (size_t)i * DMODEL * DMODEL;
    const float* Wv_i = Wv + (size_t)i * DMODEL * DMODEL;
    const float* Wp_i = Wp + (size_t)i * DMODEL * DMODEL;
    const float* bq_i = bq + (size_t)i * DMODEL;
    const float* bk_i = bk + (size_t)i * DMODEL;
    const float* bv_i = bv + (size_t)i * DMODEL;
    const float* bp_i = bp + (size_t)i * DMODEL;
    const float* W1_i = W1 + (size_t)i * DMODEL * DFF;
    const float* b1_i = b1 + (size_t)i * DFF;
    const float* W2_i = W2 + (size_t)i * DFF * DMODEL;
    const float* b2_i = b2 + (size_t)i * DMODEL;
    const float* g1_i = g1 + (size_t)i * DMODEL;
    const float* be1_i = be1 + (size_t)i * DMODEL;
    const float* g2_i = g2 + (size_t)i * DMODEL;
    const float* be2_i = be2 + (size_t)i * DMODEL;

    // Q/K/V projections: bf16 out
    gemm_wmma<float, false, 1, true><<<gProj, blk, 0, stream>>>(
        x_bf, 0, DMODEL, Wq_i, 0, DMODEL, q_bf, 0, DMODEL, bq_i, SEQ, DMODEL, DMODEL);
    gemm_wmma<float, false, 1, true><<<gProj, blk, 0, stream>>>(
        x_bf, 0, DMODEL, Wk_i, 0, DMODEL, k_bf, 0, DMODEL, bk_i, SEQ, DMODEL, DMODEL);
    gemm_wmma<float, false, 1, true><<<gProj, blk, 0, stream>>>(
        x_bf, 0, DMODEL, Wv_i, 0, DMODEL, v_bf, 0, DMODEL, bv_i, SEQ, DMODEL, DMODEL);

    // scores[h,s,t] = q_h . k_h  (+ mask), f32 out, grid.z = head (TDM A and B)
    gemm_wmma<unsigned short, true, 3, false><<<gScore, blk, 0, stream>>>(
        q_bf, DHEAD, DMODEL, k_bf, DHEAD, DMODEL,
        scores, (long)SEQ * SEQ, SEQ, nullptr, SEQ, SEQ, DHEAD);

    // row softmax, in-place f32 -> bf16
    softmax_attn_kernel<<<gSm, blk, 0, stream>>>(scores);

    // o_h = attn_h @ v_h : A = bf16 view of scores (row stride 2*SEQ elems)
    gemm_wmma<unsigned short, false, 0, true><<<gAV, blk, 0, stream>>>(
        (const unsigned short*)scores, 2L * SEQ * SEQ, 2 * SEQ,
        v_bf, DHEAD, DMODEL, oh_bf, DHEAD, DMODEL, nullptr, SEQ, DHEAD, SEQ);

    // output projection: f32 out
    gemm_wmma<float, false, 1, false><<<gProj, blk, 0, stream>>>(
        oh_bf, 0, DMODEL, Wp_i, 0, DMODEL, oproj, 0, DMODEL, bp_i, SEQ, DMODEL, DMODEL);

    // x1 = LN(oproj + x)
    ln_residual_kernel<<<SEQ, blk, 0, stream>>>(oproj, x_f, g1_i, be1_i, x1_bf, x1_f);

    // FFN
    gemm_wmma<float, false, 2, true><<<gF1, blk, 0, stream>>>(
        x1_bf, 0, DMODEL, W1_i, 0, DFF, h1_bf, 0, DFF, b1_i, SEQ, DFF, DMODEL);
    gemm_wmma<float, false, 1, false><<<gProj, blk, 0, stream>>>(
        h1_bf, 0, DFF, W2_i, 0, DMODEL, y_f, 0, DMODEL, b2_i, SEQ, DMODEL, DFF);

    // x = LN(y + x1)
    ln_residual_kernel<<<SEQ, blk, 0, stream>>>(y_f, x1_f, g2_i, be2_i, x_bf, x_f);
  }

  // logits = x @ emb^T  -> d_out (f32), then row softmax in-place
  gemm_wmma<float, true, 0, false><<<gLog, blk, 0, stream>>>(
      x_bf, 0, DMODEL, emb, 0, DMODEL, d_out, 0, VOCAB, nullptr, SEQ, VOCAB, DMODEL);
  softmax_vocab_kernel<<<SEQ, blk, 0, stream>>>((float*)d_out);
}
